// TemporalTransformer_9251359556095
// MI455X (gfx1250) — matrix operand order repaired and checked by reading, not compiled, not run on hardware
//
#include <hip/hip_runtime.h>
#include <hip/hip_bf16.h>
#include <math.h>

// ---------------------------------------------------------------------------
// TemporalTransformer for MI455X (gfx1250, wave32):
//   WMMA f16 16x16x32 GEMMs + TDM (tensor_load_to_lds) A-tile staging +
//   async global->LDS staging in attention.
// ---------------------------------------------------------------------------

typedef __attribute__((ext_vector_type(16))) _Float16 v16h;
typedef __attribute__((ext_vector_type(8)))  _Float16 v8h;
typedef __attribute__((ext_vector_type(8)))  float    v8f;
typedef __attribute__((ext_vector_type(4)))  unsigned int u32x4;
typedef __attribute__((ext_vector_type(8)))  int      i32x8;
typedef __attribute__((ext_vector_type(4)))  int      i32x4;

#define BB    4
#define CC    320
#define FF    16
#define HH    32
#define WW    32
#define INNER 320
#define MTOK  65536          // B*H*W*F tokens
#define FFINN 1280
#define NGRP  32
#define GELEM 163840         // (C/G)*F*H*W elements per (b,g)

enum { EP_BIAS = 1, EP_RES = 2, EP_OUTF = 4, EP_OUTH = 8 };

#define KB  32
#define KBP 40               // padded LDS row: 80 bytes, 16B-aligned, conflict-free

__device__ __forceinline__ v8f wmma16(v16h a, v16h b, v8f c) {
  return __builtin_amdgcn_wmma_f32_16x16x32_f16(false, a, false, b, (short)0, c,
                                                false, false);
}

__device__ __forceinline__ v16h frag2(const _Float16* p0, const _Float16* p1) {
  v8h lo = *(const v8h*)p0;
  v8h hi = *(const v8h*)p1;
  v16h r;
#pragma unroll
  for (int i = 0; i < 8; ++i) { r[i] = lo[i]; r[8 + i] = hi[i]; }
  return r;
}

// Low 32 bits of a generic pointer to __shared__ == LDS byte offset (ISA 10.2).
__device__ __forceinline__ unsigned int lds_addr32(const void* p) {
  return (unsigned int)(uintptr_t)p;
}

// ---- CDNA5 async global->LDS (ASYNCcnt) -----------------------------------
__device__ __forceinline__ void async_ld_b128(unsigned int lds, const void* gaddr) {
  asm volatile("global_load_async_to_lds_b128 %0, %1, off"
               :: "v"(lds), "v"(gaddr) : "memory");
}
__device__ __forceinline__ void wait_async0() {
#if __has_builtin(__builtin_amdgcn_s_wait_asynccnt)
  __builtin_amdgcn_s_wait_asynccnt(0);
#else
  asm volatile("s_wait_asynccnt 0x0" ::: "memory");
#endif
}

// ---- CDNA5 Tensor Data Mover: 2D tile (tile_dim0=32 elems, tile_dim1=64 rows)
// of f16 from row-major [rows x K] global into LDS with 16-byte row padding
// (pad_interval=16 DW, pad_amount=4 DW) -> matches As[64][KBP] layout exactly.
__device__ __forceinline__ void tdm_tile_64x32(const _Float16* gsrc,
                                               unsigned int lds_base, int K) {
  unsigned long long ga = (unsigned long long)(uintptr_t)gsrc;
  u32x4 g0;
  g0.x = 1u;                                    // count=1 (valid descriptor)
  g0.y = lds_base;                              // lds_addr (bytes)
  g0.z = (unsigned int)ga;                      // global_addr[31:0]
  g0.w = (unsigned int)(ga >> 32) | (2u << 30); // global_addr[56:32] | type=2
  i32x8 g1;
  g1[0] = (1 << 16) | (1 << 20) | (3 << 22) | (3 << 25);
  //       data_size=2B  pad_en   pad_int=16DW  pad_amt=4DW
  g1[1] = (int)(((unsigned)K & 0xFFFFu) << 16);          // tensor_dim0 lo16
  g1[2] = (int)(((unsigned)K >> 16) | 0u);               // dim0 hi16, dim1 lo16(=0)
  g1[3] = (int)(1u | (32u << 16));                       // dim1 hi16(=1: 65536 rows), tile_dim0=32
  g1[4] = 64;                                            // tile_dim1=64, tile_dim2=0
  g1[5] = K;                                             // tensor_dim0_stride lo32
  g1[6] = 0;
  g1[7] = 0;
  i32x4 z4 = {0, 0, 0, 0};
#if defined(__clang_major__) && (__clang_major__ >= 23)
  i32x8 z8 = {0, 0, 0, 0, 0, 0, 0, 0};
  __builtin_amdgcn_tensor_load_to_lds(g0, g1, z4, z4, z8, 0);
#else
  __builtin_amdgcn_tensor_load_to_lds(g0, g1, z4, z4, 0);
#endif
}
__device__ __forceinline__ void wait_tensor0() {
  __builtin_amdgcn_s_wait_tensorcnt(0);
}

// ---------------------------------------------------------------------------
// GroupNorm stats per (b,g) over a contiguous 163840-float region
// ---------------------------------------------------------------------------
__global__ __launch_bounds__(256)
void gn_stats_kernel(const float* __restrict__ x, float* __restrict__ stats) {
  const int bg = blockIdx.x;
  const float* p = x + (size_t)bg * GELEM;
  float s = 0.f, ss = 0.f;
  for (int i = threadIdx.x; i < GELEM; i += 256) {
    float v = p[i];
    s += v; ss += v * v;
  }
  __shared__ float rs[256], rq[256];
  rs[threadIdx.x] = s; rq[threadIdx.x] = ss;
  __syncthreads();
  for (int off = 128; off > 0; off >>= 1) {
    if (threadIdx.x < off) {
      rs[threadIdx.x] += rs[threadIdx.x + off];
      rq[threadIdx.x] += rq[threadIdx.x + off];
    }
    __syncthreads();
  }
  if (threadIdx.x == 0) {
    float mu  = rs[0] / (float)GELEM;
    float var = rq[0] / (float)GELEM - mu * mu;
    stats[2 * bg]     = mu;
    stats[2 * bg + 1] = rsqrtf(var + 1e-6f);
  }
}

__global__ __launch_bounds__(256)
void gn_apply_kernel(const float* __restrict__ x, const float* __restrict__ stats,
                     const float* __restrict__ g, const float* __restrict__ b,
                     _Float16* __restrict__ out) {
  const int idx = blockIdx.x * 256 + threadIdx.x;
  const int m = idx / CC, c = idx % CC;
  const int bb = m >> 14, h = (m >> 9) & 31, w = (m >> 4) & 31, f = m & 15;
  const int bg = bb * NGRP + c / (CC / NGRP);
  const float mu = stats[2 * bg], rstd = stats[2 * bg + 1];
  const float v = x[((((size_t)bb * CC + c) * FF + f) * HH + h) * WW + w];
  out[idx] = (_Float16)((v - mu) * rstd * g[c] + b[c]);
}

__global__ __launch_bounds__(256)
void f32_to_f16_kernel(const float* __restrict__ in, _Float16* __restrict__ out, int n) {
  int i = blockIdx.x * 256 + threadIdx.x;
  if (i < n) out[i] = (_Float16)in[i];
}

// LayerNorm over last dim (320), one wave per row, f16 out.
__global__ __launch_bounds__(256)
void ln_kernel(const float* __restrict__ t, const float* __restrict__ g,
               const float* __restrict__ b, _Float16* __restrict__ out) {
  const int lane = threadIdx.x & 31;
  const int row  = blockIdx.x * 8 + (threadIdx.x >> 5);
  const float* p = t + (size_t)row * INNER;
  float vals[10];
  float s = 0.f, ss = 0.f;
#pragma unroll
  for (int i = 0; i < 10; ++i) {
    float v = p[lane + i * 32];
    vals[i] = v; s += v; ss += v * v;
  }
#pragma unroll
  for (int msk = 1; msk < 32; msk <<= 1) {
    s  += __shfl_xor(s, msk);
    ss += __shfl_xor(ss, msk);
  }
  const float mu = s / (float)INNER;
  const float rstd = rsqrtf(ss / (float)INNER - mu * mu + 1e-5f);
#pragma unroll
  for (int i = 0; i < 10; ++i) {
    int c = lane + i * 32;
    out[(size_t)row * INNER + c] = (_Float16)((vals[i] - mu) * rstd * g[c] + b[c]);
  }
}

// ---------------------------------------------------------------------------
// WMMA GEMM, A tile staged by the Tensor Data Mover.
// Block 256 = 8 waves; 64x64 tile; wave = 1 m-tile x 2 n-subtiles.
// ---------------------------------------------------------------------------
template<int EP>
__global__ __launch_bounds__(256)
void gemm_f16(const _Float16* __restrict__ A, const _Float16* __restrict__ Wt,
              const float* __restrict__ bias, const float* res,
              float* outf, _Float16* outh, int N, int K) {
  __shared__ _Float16 As[64][KBP];
  __shared__ _Float16 Bs[64][KBP];
  const int m0 = blockIdx.x * 64, n0 = blockIdx.y * 64;
  const int tid = threadIdx.x;
  const int lane = tid & 31;
  const int l16 = lane & 15, half = lane >> 4;
  const int mt = (tid >> 5) & 3;
  const int ng = tid >> 7;
  const int bn = tid & 63, bk = (tid >> 6) * 8;   // B staging: 8 f16/thread
  const unsigned int as_base = lds_addr32(&As[0][0]);
  v8f acc[2] = {{}, {}};

  for (int k0 = 0; k0 < K; k0 += KB) {
    if (tid < 32)  // one TDM descriptor per block per K-step (wave 0)
      tdm_tile_64x32(A + (size_t)m0 * K + k0, as_base, K);
#pragma unroll
    for (int i = 0; i < 8; ++i)
      Bs[bn][bk + i] = Wt[(size_t)(k0 + bk + i) * N + n0 + bn];
    if (k0 + KB < K)  // prefetch next B rows -> global_prefetch_b8
      __builtin_prefetch(Wt + (size_t)(k0 + KB + bk) * N + n0 + bn, 0, 3);
    wait_tensor0();
    __syncthreads();
    v16h af  = frag2(&As[mt * 16 + l16][half * 16],
                     &As[mt * 16 + l16][half * 16 + 8]);
    v16h bf0 = frag2(&Bs[ng * 32 + l16][half * 16],
                     &Bs[ng * 32 + l16][half * 16 + 8]);
    v16h bf1 = frag2(&Bs[ng * 32 + 16 + l16][half * 16],
                     &Bs[ng * 32 + 16 + l16][half * 16 + 8]);
    acc[0] = wmma16(af, bf0, acc[0]);
    acc[1] = wmma16(af, bf1, acc[1]);
    __syncthreads();
  }

  const int mrow = m0 + mt * 16 + half * 8;
#pragma unroll
  for (int s = 0; s < 2; ++s) {
    const int n = n0 + ng * 32 + s * 16 + l16;
    const float bv = (EP & EP_BIAS) ? bias[n] : 0.0f;
#pragma unroll
    for (int r = 0; r < 8; ++r) {
      const size_t o = (size_t)(mrow + r) * N + n;
      float v = acc[s][r] + bv;
      if (EP & EP_RES)  v += res[o];
      if (EP & EP_OUTF) outf[o] = v;
      if (EP & EP_OUTH) outh[o] = (_Float16)v;
    }
  }
}

// ---------------------------------------------------------------------------
// GEGLU GEMM: both halves of W1, fused a * gelu(g) -> f16
// ---------------------------------------------------------------------------
__global__ __launch_bounds__(256)
void gemm_geglu(const _Float16* __restrict__ A, const _Float16* __restrict__ W1,
                const float* __restrict__ b1, _Float16* __restrict__ out) {
  __shared__ _Float16 As[64][KBP];
  __shared__ _Float16 Ba[32][KBP];
  __shared__ _Float16 Bg[32][KBP];
  const int m0 = blockIdx.x * 64, n0 = blockIdx.y * 32;
  const int tid = threadIdx.x;
  const int lane = tid & 31;
  const int l16 = lane & 15, half = lane >> 4;
  const int mt = (tid >> 5) & 3, nt = tid >> 7;
  const int bn = tid & 31, bq = (tid >> 5) * 4;
  const unsigned int as_base = lds_addr32(&As[0][0]);
  v8f acca = {}, accg = {};

  for (int k0 = 0; k0 < INNER; k0 += KB) {
    if (tid < 32)
      tdm_tile_64x32(A + (size_t)m0 * INNER + k0, as_base, INNER);
#pragma unroll
    for (int i = 0; i < 4; ++i) {
      const size_t krow = (size_t)(k0 + bq + i) * (2 * FFINN);
      Ba[bn][bq + i] = W1[krow + n0 + bn];
      Bg[bn][bq + i] = W1[krow + FFINN + n0 + bn];
    }
    wait_tensor0();
    __syncthreads();
    v16h af  = frag2(&As[mt * 16 + l16][half * 16],
                     &As[mt * 16 + l16][half * 16 + 8]);
    v16h bfa = frag2(&Ba[nt * 16 + l16][half * 16],
                     &Ba[nt * 16 + l16][half * 16 + 8]);
    v16h bfg = frag2(&Bg[nt * 16 + l16][half * 16],
                     &Bg[nt * 16 + l16][half * 16 + 8]);
    acca = wmma16(af, bfa, acca);
    accg = wmma16(af, bfg, accg);
    __syncthreads();
  }

  const int n = n0 + nt * 16 + l16;
  const int mrow = m0 + mt * 16 + half * 8;
  const float ba = b1[n], bgv = b1[FFINN + n];
#pragma unroll
  for (int r = 0; r < 8; ++r) {
    float a = acca[r] + ba;
    float g = accg[r] + bgv;
    float ge = 0.5f * g * (1.0f + erff(g * 0.70710678118654752f));
    out[(size_t)(mrow + r) * FFINN + n] = (_Float16)(a * ge);
  }
}

// ---------------------------------------------------------------------------
// proj_out GEMM with scatter epilogue: out[b,c,f,h,w] = acc + bias + x_in
// ---------------------------------------------------------------------------
__global__ __launch_bounds__(256)
void gemm_scatter(const _Float16* __restrict__ A, const _Float16* __restrict__ Wt,
                  const float* __restrict__ bias, const float* __restrict__ xin,
                  float* __restrict__ out) {
  __shared__ _Float16 As[64][KBP];
  __shared__ _Float16 Bs[64][KBP];
  const int m0 = blockIdx.x * 64, n0 = blockIdx.y * 64;
  const int tid = threadIdx.x;
  const int lane = tid & 31;
  const int l16 = lane & 15, half = lane >> 4;
  const int mt = (tid >> 5) & 3, ng = tid >> 7;
  const int bn = tid & 63, bk = (tid >> 6) * 8;
  const unsigned int as_base = lds_addr32(&As[0][0]);
  v8f acc[2] = {{}, {}};

  for (int k0 = 0; k0 < INNER; k0 += KB) {
    if (tid < 32)
      tdm_tile_64x32(A + (size_t)m0 * INNER + k0, as_base, INNER);
#pragma unroll
    for (int i = 0; i < 8; ++i)
      Bs[bn][bk + i] = Wt[(size_t)(k0 + bk + i) * CC + n0 + bn];
    wait_tensor0();
    __syncthreads();
    v16h af  = frag2(&As[mt * 16 + l16][half * 16],
                     &As[mt * 16 + l16][half * 16 + 8]);
    v16h bf0 = frag2(&Bs[ng * 32 + l16][half * 16],
                     &Bs[ng * 32 + l16][half * 16 + 8]);
    v16h bf1 = frag2(&Bs[ng * 32 + 16 + l16][half * 16],
                     &Bs[ng * 32 + 16 + l16][half * 16 + 8]);
    acc[0] = wmma16(af, bf0, acc[0]);
    acc[1] = wmma16(af, bf1, acc[1]);
    __syncthreads();
  }

  const int mrow = m0 + mt * 16 + half * 8;
#pragma unroll
  for (int s = 0; s < 2; ++s) {
    const int c = n0 + ng * 32 + s * 16 + l16;
    const float bv = bias[c];
#pragma unroll
    for (int r = 0; r < 8; ++r) {
      const int m = mrow + r;
      const int bb = m >> 14, h = (m >> 9) & 31, w = (m >> 4) & 31, f = m & 15;
      const size_t o = ((((size_t)bb * CC + c) * FF + f) * HH + h) * WW + w;
      out[o] = acc[s][r] + bv + xin[o];
    }
  }
}

// ---------------------------------------------------------------------------
// Attention: one wave per (token-batch, head). Q/K staged via async
// global->LDS (ASYNCcnt); V transposed manually; 2 + 3 WMMAs.
// ---------------------------------------------------------------------------
__global__ __launch_bounds__(32)
void attn_kernel(const _Float16* __restrict__ Q, const _Float16* __restrict__ Kx,
                 const _Float16* __restrict__ V, _Float16* __restrict__ O) {
  __shared__ _Float16 Qp[16][72];   // 16 tok x 64 dims (144B rows)
  __shared__ _Float16 Kp[16][72];
  __shared__ _Float16 Vt[48][40];   // dims(pad 48) x tok(pad 32)
  __shared__ _Float16 Pl[16][40];
  const int bt = blockIdx.x;
  const int hd = blockIdx.y * 40;
  const int lane = threadIdx.x;
  const int l16 = lane & 15, half = lane >> 4;
  const size_t rb = (size_t)bt * 16 * INNER;

  // async-copy the real 40-dim rows (5 x 16B chunks per row) straight to LDS
  for (int c = lane; c < 80; c += 32) {
    const int tok = c / 5, seg = (c % 5) * 8;
    const size_t gofs = rb + (size_t)tok * INNER + hd + seg;
    async_ld_b128(lds_addr32(&Qp[tok][seg]), Q + gofs);
    async_ld_b128(lds_addr32(&Kp[tok][seg]), Kx + gofs);
  }
  // zero only the pad columns (disjoint from async destinations)
  for (int idx = lane; idx < 16 * 24; idx += 32) {
    const int tok = idx / 24, d = 40 + idx % 24;
    Qp[tok][d] = (_Float16)0.f;
    Kp[tok][d] = (_Float16)0.f;
  }
  for (int idx = lane; idx < 48 * 32; idx += 32) {
    const int d = idx >> 5, tok = idx & 31;
    _Float16 vv = (_Float16)0.f;
    if (d < 40 && tok < 16) vv = V[rb + (size_t)tok * INNER + hd + d];
    Vt[d][tok] = vv;
  }
  wait_async0();
  __syncthreads();

  v8f sim = {};
#pragma unroll
  for (int k0 = 0; k0 < 64; k0 += 32) {
    v16h af = frag2(&Qp[l16][k0 + half * 16], &Qp[l16][k0 + half * 16 + 8]);
    v16h bf = frag2(&Kp[l16][k0 + half * 16], &Kp[l16][k0 + half * 16 + 8]);
    sim = wmma16(af, bf, sim);
  }

#pragma unroll
  for (int r = 0; r < 8; ++r) {
    float s = sim[r] * 0.15811388300841898f;      // DHEAD^-0.5
    float mx = s;
#pragma unroll
    for (int msk = 1; msk < 16; msk <<= 1) mx = fmaxf(mx, __shfl_xor(mx, msk));
    float e = __expf(s - mx);
    float sum = e;
#pragma unroll
    for (int msk = 1; msk < 16; msk <<= 1) sum += __shfl_xor(sum, msk);
    const float p = e / sum;
    Pl[half * 8 + r][l16]      = (_Float16)p;
    Pl[half * 8 + r][16 + l16] = (_Float16)0.f;
  }
  __syncthreads();

#pragma unroll
  for (int nt = 0; nt < 3; ++nt) {
    v8f o = {};
    v16h af = frag2(&Pl[l16][half * 16], &Pl[l16][half * 16 + 8]);
    v16h bf = frag2(&Vt[nt * 16 + l16][half * 16],
                    &Vt[nt * 16 + l16][half * 16 + 8]);
    o = wmma16(af, bf, o);
    const int d = nt * 16 + l16;
#pragma unroll
    for (int r = 0; r < 8; ++r) {
      if (d < 40)
        O[rb + (size_t)(half * 8 + r) * INNER + hd + d] = (_Float16)o[r];
    }
  }
}

// ---------------------------------------------------------------------------
// Host launcher
// ---------------------------------------------------------------------------
extern "C" void kernel_launch(void* const* d_in, const int* in_sizes, int n_in,
                              void* d_out, int out_size, void* d_ws, size_t ws_size,
                              hipStream_t stream) {
  (void)in_sizes; (void)n_in; (void)out_size; (void)ws_size;
  const float* x         = (const float*)d_in[0];
  const float* gn_g      = (const float*)d_in[1];
  const float* gn_b      = (const float*)d_in[2];
  const float* proj_in_w = (const float*)d_in[3];
  const float* proj_in_b = (const float*)d_in[4];
  const float* ln1_g = (const float*)d_in[5];
  const float* ln1_b = (const float*)d_in[6];
  const float* wq1 = (const float*)d_in[7];
  const float* wk1 = (const float*)d_in[8];
  const float* wv1 = (const float*)d_in[9];
  const float* wo1 = (const float*)d_in[10];
  const float* bo1 = (const float*)d_in[11];
  const float* ln2_g = (const float*)d_in[12];
  const float* ln2_b = (const float*)d_in[13];
  const float* wq2 = (const float*)d_in[14];
  const float* wk2 = (const float*)d_in[15];
  const float* wv2 = (const float*)d_in[16];
  const float* wo2 = (const float*)d_in[17];
  const float* bo2 = (const float*)d_in[18];
  const float* ln3_g = (const float*)d_in[19];
  const float* ln3_b = (const float*)d_in[20];
  const float* ff_w1 = (const float*)d_in[21];
  const float* ff_b1 = (const float*)d_in[22];
  const float* ff_w2 = (const float*)d_in[23];
  const float* ff_b2 = (const float*)d_in[24];
  const float* proj_out_w = (const float*)d_in[25];
  const float* proj_out_b = (const float*)d_in[26];
  float* out = (float*)d_out;

  char* ws = (char*)d_ws;
  size_t off = 0;
  auto alloc = [&](size_t bytes) -> void* {
    void* p = ws + off;
    off += (bytes + 255) & ~(size_t)255;
    return p;
  };
  float*     t     = (float*)    alloc((size_t)MTOK * INNER * 4);
  _Float16*  xln   = (_Float16*) alloc((size_t)MTOK * INNER * 2);
  _Float16*  qh    = (_Float16*) alloc((size_t)MTOK * INNER * 2);
  _Float16*  kh    = (_Float16*) alloc((size_t)MTOK * INNER * 2);
  _Float16*  vh    = (_Float16*) alloc((size_t)MTOK * INNER * 2);
  _Float16*  oh    = (_Float16*) alloc((size_t)MTOK * INNER * 2);
  _Float16*  gg    = (_Float16*) alloc((size_t)MTOK * FFINN * 2);
  _Float16*  w_pi  = (_Float16*) alloc((size_t)INNER * INNER * 2);
  _Float16*  w_q1  = (_Float16*) alloc((size_t)INNER * INNER * 2);
  _Float16*  w_k1  = (_Float16*) alloc((size_t)INNER * INNER * 2);
  _Float16*  w_v1  = (_Float16*) alloc((size_t)INNER * INNER * 2);
  _Float16*  w_o1  = (_Float16*) alloc((size_t)INNER * INNER * 2);
  _Float16*  w_q2  = (_Float16*) alloc((size_t)INNER * INNER * 2);
  _Float16*  w_k2  = (_Float16*) alloc((size_t)INNER * INNER * 2);
  _Float16*  w_v2  = (_Float16*) alloc((size_t)INNER * INNER * 2);
  _Float16*  w_o2  = (_Float16*) alloc((size_t)INNER * INNER * 2);
  _Float16*  w_f1  = (_Float16*) alloc((size_t)INNER * 2 * FFINN * 2);
  _Float16*  w_f2  = (_Float16*) alloc((size_t)FFINN * INNER * 2);
  _Float16*  w_po  = (_Float16*) alloc((size_t)INNER * INNER * 2);
  float*     stats = (float*)    alloc(2 * BB * NGRP * 4);

  const dim3 blk(256);
  const dim3 gemm_grid(MTOK / 64, INNER / 64);
  const dim3 geglu_grid(MTOK / 64, FFINN / 32);
  const dim3 attn_grid(MTOK / 16, 8);
  const int  sq  = INNER * INNER;
  const int  cgr = (sq + 255) / 256;

  f32_to_f16_kernel<<<cgr, blk, 0, stream>>>(proj_in_w, w_pi, sq);
  f32_to_f16_kernel<<<cgr, blk, 0, stream>>>(wq1, w_q1, sq);
  f32_to_f16_kernel<<<cgr, blk, 0, stream>>>(wk1, w_k1, sq);
  f32_to_f16_kernel<<<cgr, blk, 0, stream>>>(wv1, w_v1, sq);
  f32_to_f16_kernel<<<cgr, blk, 0, stream>>>(wo1, w_o1, sq);
  f32_to_f16_kernel<<<cgr, blk, 0, stream>>>(wq2, w_q2, sq);
  f32_to_f16_kernel<<<cgr, blk, 0, stream>>>(wk2, w_k2, sq);
  f32_to_f16_kernel<<<cgr, blk, 0, stream>>>(wv2, w_v2, sq);
  f32_to_f16_kernel<<<cgr, blk, 0, stream>>>(wo2, w_o2, sq);
  f32_to_f16_kernel<<<(INNER * 2 * FFINN + 255) / 256, blk, 0, stream>>>(ff_w1, w_f1, INNER * 2 * FFINN);
  f32_to_f16_kernel<<<(FFINN * INNER + 255) / 256, blk, 0, stream>>>(ff_w2, w_f2, FFINN * INNER);
  f32_to_f16_kernel<<<cgr, blk, 0, stream>>>(proj_out_w, w_po, sq);

  gn_stats_kernel<<<BB * NGRP, blk, 0, stream>>>(x, stats);
  gn_apply_kernel<<<(MTOK * INNER) / 256, blk, 0, stream>>>(x, stats, gn_g, gn_b, xln);

  gemm_f16<EP_BIAS | EP_OUTF><<<gemm_grid, blk, 0, stream>>>(
      xln, w_pi, proj_in_b, nullptr, t, nullptr, INNER, INNER);

  // ---- attention block 1 ----
  ln_kernel<<<MTOK / 8, blk, 0, stream>>>(t, ln1_g, ln1_b, xln);
  gemm_f16<EP_OUTH><<<gemm_grid, blk, 0, stream>>>(xln, w_q1, nullptr, nullptr, nullptr, qh, INNER, INNER);
  gemm_f16<EP_OUTH><<<gemm_grid, blk, 0, stream>>>(xln, w_k1, nullptr, nullptr, nullptr, kh, INNER, INNER);
  gemm_f16<EP_OUTH><<<gemm_grid, blk, 0, stream>>>(xln, w_v1, nullptr, nullptr, nullptr, vh, INNER, INNER);
  attn_kernel<<<attn_grid, dim3(32), 0, stream>>>(qh, kh, vh, oh);
  gemm_f16<EP_BIAS | EP_RES | EP_OUTF><<<gemm_grid, blk, 0, stream>>>(
      oh, w_o1, bo1, t, t, nullptr, INNER, INNER);

  // ---- attention block 2 ----
  ln_kernel<<<MTOK / 8, blk, 0, stream>>>(t, ln2_g, ln2_b, xln);
  gemm_f16<EP_OUTH><<<gemm_grid, blk, 0, stream>>>(xln, w_q2, nullptr, nullptr, nullptr, qh, INNER, INNER);
  gemm_f16<EP_OUTH><<<gemm_grid, blk, 0, stream>>>(xln, w_k2, nullptr, nullptr, nullptr, kh, INNER, INNER);
  gemm_f16<EP_OUTH><<<gemm_grid, blk, 0, stream>>>(xln, w_v2, nullptr, nullptr, nullptr, vh, INNER, INNER);
  attn_kernel<<<attn_grid, dim3(32), 0, stream>>>(qh, kh, vh, oh);
  gemm_f16<EP_BIAS | EP_RES | EP_OUTF><<<gemm_grid, blk, 0, stream>>>(
      oh, w_o2, bo2, t, t, nullptr, INNER, INNER);

  // ---- GEGLU feed-forward ----
  ln_kernel<<<MTOK / 8, blk, 0, stream>>>(t, ln3_g, ln3_b, xln);
  gemm_geglu<<<geglu_grid, blk, 0, stream>>>(xln, w_f1, ff_b1, gg);
  gemm_f16<EP_BIAS | EP_RES | EP_OUTF><<<gemm_grid, blk, 0, stream>>>(
      gg, w_f2, ff_b2, t, t, nullptr, INNER, FFINN);

  // ---- proj_out + input residual ----
  f32_to_f16_kernel<<<(MTOK * INNER + 255) / 256, blk, 0, stream>>>(t, xln, MTOK * INNER);
  gemm_scatter<<<gemm_grid, blk, 0, stream>>>(xln, w_po, proj_out_b, x, out);
}
